// Network_43361989821084
// MI455X (gfx1250) — matrix operand order repaired and checked by reading, not compiled, not run on hardware
//
#include <hip/hip_runtime.h>
#include <math.h>

typedef __bf16 bf16;
typedef __attribute__((ext_vector_type(16))) __bf16 v16bf;
typedef __attribute__((ext_vector_type(8)))  float  v8f;

#define NPIL 8192
#define NCR  128
#define HID  512
#define NH   4
#define DH   128
#define BATCH 128
#define TT   8
#define NCOS 64
#define ACT  64
#define DEG  16
#define PILES_PER 64

#define EPI_SELU 1
#define EPI_GATE 2

__device__ __forceinline__ float selu_f(float x) {
    const float lam = 1.0507009873554805f, alp = 1.6732632423543772f;
    return x > 0.f ? lam * x : lam * alp * (expf(x) - 1.f);
}
__device__ __forceinline__ float gelu_f(float x) {
    return 0.5f * x * (1.f + erff(x * 0.70710678118654752f));
}

union FragU { uint4 u[2]; v16bf v; };

// ---------------------------------------------------------------------------
// bf16 WMMA GEMM with fragment-packed B operand.
// B pre-packed per (ktile=32 x ntile=16) tile in per-lane fragment order:
//   elem index = (((kt*bNtiles + nt)*32 + lane)*16 + e)
// A row-major bf16 (lda multiple of 8 -> 16B-aligned 2x uint4 loads).
// Block = 128 threads (4 waves); wave computes a 16x64 strip.
// Per K-step: issue ALL fragment loads (2 A + 8 B b128), then 4 WMMAs, so
// the scheduler can use partial s_wait_loadcnt instead of full drains.
// Grid: (N/64, M/64); K multiple of 32.
// Epilogue: +bias -> sigmoid-gated residual -> SELU -> dual f32/bf16 store.
// ---------------------------------------------------------------------------
__global__ __launch_bounds__(128) void gemm_bf16_wmma(
    const bf16* __restrict__ A, int lda,
    const bf16* __restrict__ Bp, int bNtiles,
    const float* __restrict__ bias,
    float* __restrict__ Cf, bf16* __restrict__ Cbf, int ldc,
    int K, int flags, const float* __restrict__ gatep,
    const float* __restrict__ Res, int ldr)
{
    const int lane = threadIdx.x & 31;
    const int wave = threadIdx.x >> 5;
    const int m0 = blockIdx.y * 64 + wave * 16;
    const int n0 = blockIdx.x * 64;
    const int nt0 = blockIdx.x * 4;
    const int mrow = lane & 15;
    const int half = lane >> 4;          // 0 or 1 (K-half selector)

    v8f zero = {0.f,0.f,0.f,0.f,0.f,0.f,0.f,0.f};
    v8f acc[4] = {zero, zero, zero, zero};

    const bf16* Arow = A + (long)(m0 + mrow) * lda;

    for (int k0 = 0; k0 < K; k0 += 32) {
        const int kt = k0 >> 5;
        // ---- load phase: 2x b128 (A) + 8x b128 (B tiles) ----
        FragU ua;
        const bf16* ap = Arow + k0 + half * 16;
        ua.u[0] = *(const uint4*)(ap);
        ua.u[1] = *(const uint4*)(ap + 8);
        FragU ub[4];
        const bf16* bbase = Bp + (((long)kt * bNtiles + nt0) * 32 + lane) * 16;
#pragma unroll
        for (int j = 0; j < 4; ++j) {
            const bf16* bp = bbase + (long)j * 32 * 16;
            ub[j].u[0] = *(const uint4*)(bp);
            ub[j].u[1] = *(const uint4*)(bp + 8);
        }
        if (k0 + 32 < K) __builtin_prefetch(Arow + k0 + 32, 0, 0);
        // ---- compute phase: 4 WMMAs, waits can retire incrementally ----
#pragma unroll
        for (int j = 0; j < 4; ++j)
            acc[j] = __builtin_amdgcn_wmma_f32_16x16x32_bf16(
                false, ua.v, false, ub[j].v, (short)0, acc[j], false, false);
    }

    float g = 1.f, go = 0.f;
    if (flags & EPI_GATE) {
        float s = *gatep;
        g = 1.f / (1.f + expf(-s));
        go = 1.f - g;
    }
#pragma unroll
    for (int j = 0; j < 4; ++j) {
        int col = n0 + j * 16 + mrow;
        float bv = bias ? bias[col] : 0.f;
#pragma unroll
        for (int r = 0; r < 8; ++r) {
            int row = m0 + r + half * 8;       // C/D layout: VGPR r -> M=r(+8)
            float c = acc[j][r] + bv;
            if (flags & EPI_GATE) c = g * c + go * Res[(long)row * ldr + col];
            if (flags & EPI_SELU) c = selu_f(c);
            if (Cf)  Cf[(long)row * ldc + col]  = c;
            if (Cbf) Cbf[(long)row * ldc + col] = (bf16)c;
        }
    }
}

// ---------------------------------------------------------------------------
// f32 [srcR x srcC] -> packed-fragment bf16 [Kd x Nd] (zero pad OOB).
// ---------------------------------------------------------------------------
__global__ void pack_b(const float* __restrict__ src, bf16* __restrict__ dst,
                       int srcR, int srcC, int Nd, long n)
{
    long i = (long)blockIdx.x * blockDim.x + threadIdx.x;
    if (i >= n) return;
    int e = (int)(i & 15);
    int lane = (int)((i >> 4) & 31);
    long tile = i >> 9;
    int ntiles = Nd >> 4;
    int nt = (int)(tile % ntiles);
    int kt = (int)(tile / ntiles);
    int col = nt * 16 + (lane & 15);
    int k = kt * 32 + (lane >> 4) * 16 + e;
    float v = (k < srcR && col < srcC) ? src[(long)k * srcC + col] : 0.f;
    dst[i] = (bf16)v;
}

// f32 -> bf16 row-major convert with zero padding (for A-side inputs)
__global__ void pad_cvt(const float* __restrict__ src, bf16* __restrict__ dst,
                        int srcR, int srcC, int dstC, long n)
{
    long i = (long)blockIdx.x * blockDim.x + threadIdx.x;
    if (i >= n) return;
    long r = i / dstC; long c = i - r * dstC;
    float v = (r < srcR && c < srcC) ? src[r * (long)srcC + c] : 0.f;
    dst[i] = (bf16)v;
}

__global__ void gelu_cvt(const float* __restrict__ src, bf16* __restrict__ dst, long n)
{
    long i = (long)blockIdx.x * blockDim.x + threadIdx.x;
    if (i >= n) return;
    dst[i] = (bf16)gelu_f(src[i]);
}

// ---------------------------------------------------------------------------
// p2c edges: dst crane g receives from piles g*64..g*64+63.
// ---------------------------------------------------------------------------
__global__ __launch_bounds__(128) void edge_p2c(
    const float* __restrict__ Qc, const bf16* __restrict__ KA,
    const bf16* __restrict__ VM, const float* __restrict__ prel,
    float* __restrict__ AGGc)
{
    int g = blockIdx.x, h = blockIdx.y, tid = threadIdx.x;
    __shared__ float sa[PILES_PER];
    const float scale = 0.08838834764831845f;   // 1/sqrt(128)
    if (tid < PILES_PER) {
        int src = g * PILES_PER + tid;
        const float* q  = Qc + (long)g * HID + h * DH;
        const bf16*  ka = KA + (long)src * HID + h * DH;
        float dot = 0.f;
        for (int f = 0; f < DH; ++f) dot += q[f] * (float)ka[f];
        sa[tid] = dot * prel[h] * scale;
    }
    __syncthreads();
    if (tid == 0) {
        float m = sa[0];
        for (int j = 1; j < PILES_PER; ++j) m = fmaxf(m, sa[j]);
        float s = 0.f;
        for (int j = 0; j < PILES_PER; ++j) { float e = expf(sa[j] - m); sa[j] = e; s += e; }
        float inv = 1.f / (s + 1e-16f);
        for (int j = 0; j < PILES_PER; ++j) sa[j] *= inv;
    }
    __syncthreads();
    int f = tid;   // 0..127
    float acc = 0.f;
    for (int j = 0; j < PILES_PER; ++j)
        acc += sa[j] * (float)VM[((long)(g * PILES_PER + j)) * HID + h * DH + f];
    AGGc[(long)g * HID + h * DH + f] = acc;
}

// ---------------------------------------------------------------------------
// pile aggregation: p2p (16 contiguous incoming edges) softmax + c2p (single
// incoming edge -> att == 1).
// ---------------------------------------------------------------------------
__global__ __launch_bounds__(128) void edge_pile(
    const float* __restrict__ Qp, const bf16* __restrict__ KApp,
    const bf16* __restrict__ VMpp, const bf16* __restrict__ VMcp,
    const int* __restrict__ src_pp, const float* __restrict__ prel_pp,
    float* __restrict__ AGGp)
{
    int i = blockIdx.x, h = blockIdx.y, tid = threadIdx.x;
    __shared__ float sa[DEG];
    __shared__ int   ssrc[DEG];
    const float scale = 0.08838834764831845f;
    if (tid < DEG) {
        int src = src_pp[(long)i * DEG + tid];
        ssrc[tid] = src;
        const float* q  = Qp + (long)i * HID + h * DH;
        const bf16*  ka = KApp + (long)src * HID + h * DH;
        float dot = 0.f;
        for (int f = 0; f < DH; ++f) dot += q[f] * (float)ka[f];
        sa[tid] = dot * prel_pp[h] * scale;
    }
    __syncthreads();
    if (tid == 0) {
        float m = sa[0];
        for (int j = 1; j < DEG; ++j) m = fmaxf(m, sa[j]);
        float s = 0.f;
        for (int j = 0; j < DEG; ++j) { float e = expf(sa[j] - m); sa[j] = e; s += e; }
        float inv = 1.f / (s + 1e-16f);
        for (int j = 0; j < DEG; ++j) sa[j] *= inv;
    }
    __syncthreads();
    int f = tid;
    int g = i >> 6;                                  // pile's graph/crane
    float acc = (float)VMcp[(long)g * HID + h * DH + f];   // c2p, att == 1
    for (int j = 0; j < DEG; ++j)
        acc += sa[j] * (float)VMpp[(long)ssrc[j] * HID + h * DH + f];
    AGGp[(long)i * HID + h * DH + f] = acc;
}

// pooled = sum over 64 piles; h = crane + pooled
__global__ __launch_bounds__(128) void pool_combine(
    const float* __restrict__ Op, const float* __restrict__ Oc,
    float* __restrict__ hvec)
{
    int b = blockIdx.x;
    int f = blockIdx.y * 128 + threadIdx.x;
    float s = 0.f;
    for (int p = 0; p < PILES_PER; ++p)
        s += Op[((long)(b * PILES_PER + p)) * HID + f];
    hvec[(long)b * HID + f] = Oc[(long)b * HID + f] + s;
}

// cos features: A[i, j] = cos(taus[i] * pi * (j+1))
__global__ void cos_feat(const float* __restrict__ taus, bf16* __restrict__ A)
{
    int i = blockIdx.x, j = threadIdx.x;
    A[(long)i * NCOS + j] = (bf16)cosf(taus[i] * 3.14159265358979323846f * (float)(j + 1));
}

// z[i,f] = h[i/T, f] * cosx[i,f]  (-> bf16)
__global__ void mul_hz(const float* __restrict__ hvec, const float* __restrict__ cosx,
                       bf16* __restrict__ zb, long n)
{
    long i = (long)blockIdx.x * blockDim.x + threadIdx.x;
    if (i >= n) return;
    long row = i >> 9;            // /HID
    long b = row >> 3;            // /T
    long f = i & (HID - 1);
    zb[i] = (bf16)(hvec[b * HID + f] * cosx[i]);
}

__global__ void copy_f32(const float* __restrict__ src, float* __restrict__ dst, int n)
{
    int i = blockIdx.x * blockDim.x + threadIdx.x;
    if (i < n) dst[i] = src[i];
}

// ===========================================================================
extern "C" void kernel_launch(void* const* d_in, const int* in_sizes, int n_in,
                              void* d_out, int out_size, void* d_ws, size_t ws_size,
                              hipStream_t stream)
{
    (void)in_sizes; (void)n_in; (void)out_size; (void)ws_size;
    auto F = [&](int i) { return (const float*)d_in[i]; };
    const int* ei_pp = (const int*)d_in[4];      // [2, NP*DEG]; src row first
    const float* taus = F(5);
    float* outq = (float*)d_out;

    // ------- bump allocator over d_ws -------
    char* ws = (char*)d_ws;
    size_t off = 0;
    auto alloc = [&](size_t n, size_t esz) -> void* {
        off = (off + 255) & ~(size_t)255;
        void* p = ws + off;
        off += n * esz;
        return p;
    };

    // packed bf16 weights (B operands)
    bf16 *Wk[2][2], *Wq[2][2], *Wv[2][2], *Wa[2][2];
    bf16 *Arel[2][3], *Mrel[2][3];
    for (int L = 0; L < 2; ++L) {
        int KL = L ? HID : 96;
        for (int t = 0; t < 2; ++t) {
            Wk[L][t] = (bf16*)alloc((size_t)KL * HID, 2);
            Wq[L][t] = (bf16*)alloc((size_t)KL * HID, 2);
            Wv[L][t] = (bf16*)alloc((size_t)KL * HID, 2);
            Wa[L][t] = (bf16*)alloc((size_t)HID * HID, 2);
        }
        for (int r = 0; r < 3; ++r) {
            Arel[L][r] = (bf16*)alloc((size_t)NH * DH * DH, 2);
            Mrel[L][r] = (bf16*)alloc((size_t)NH * DH * DH, 2);
        }
    }
    bf16* cosWb = (bf16*)alloc((size_t)NCOS * HID, 2);
    bf16* ff1Wb = (bf16*)alloc((size_t)HID * HID, 2);
    bf16* ff2Wb = (bf16*)alloc((size_t)HID * ACT, 2);

    // bf16 activations (A operands, row-major)
    bf16* Xc[2]; bf16* Xp[2];
    Xc[0] = (bf16*)alloc((size_t)NCR * 96, 2);
    Xp[0] = (bf16*)alloc((size_t)NPIL * 96, 2);
    Xc[1] = (bf16*)alloc((size_t)NCR * HID, 2);
    Xp[1] = (bf16*)alloc((size_t)NPIL * HID, 2);
    bf16* Kcb = (bf16*)alloc((size_t)NCR * HID, 2);
    bf16* Vcb = (bf16*)alloc((size_t)NCR * HID, 2);
    bf16* Kpb = (bf16*)alloc((size_t)NPIL * HID, 2);
    bf16* Vpb = (bf16*)alloc((size_t)NPIL * HID, 2);
    bf16* KAcp = (bf16*)alloc((size_t)NCR * HID, 2);
    bf16* VMcp = (bf16*)alloc((size_t)NCR * HID, 2);
    bf16* KApc = (bf16*)alloc((size_t)NPIL * HID, 2);
    bf16* VMpc = (bf16*)alloc((size_t)NPIL * HID, 2);
    bf16* KApp = (bf16*)alloc((size_t)NPIL * HID, 2);
    bf16* VMpp = (bf16*)alloc((size_t)NPIL * HID, 2);
    bf16* Gc = (bf16*)alloc((size_t)NCR * HID, 2);
    bf16* Gp = (bf16*)alloc((size_t)NPIL * HID, 2);
    bf16* cosA = (bf16*)alloc((size_t)BATCH * TT * NCOS, 2);
    bf16* zb1 = (bf16*)alloc((size_t)BATCH * TT * HID, 2);
    bf16* zb2 = (bf16*)alloc((size_t)BATCH * TT * HID, 2);

    // f32 buffers
    float* Qc   = (float*)alloc((size_t)NCR * HID, 4);
    float* Qp   = (float*)alloc((size_t)NPIL * HID, 4);
    float* AGGc = (float*)alloc((size_t)NCR * HID, 4);
    float* AGGp = (float*)alloc((size_t)NPIL * HID, 4);
    float* O1c  = (float*)alloc((size_t)NCR * HID, 4);
    float* O1p  = (float*)alloc((size_t)NPIL * HID, 4);
    float* O2c  = (float*)alloc((size_t)NCR * HID, 4);
    float* O2p  = (float*)alloc((size_t)NPIL * HID, 4);
    float* hvec = (float*)alloc((size_t)BATCH * HID, 4);
    float* cosx = (float*)alloc((size_t)BATCH * TT * HID, 4);

    auto cvt = [&](const float* s, bf16* d, int srcR, int srcC, int dstC, long n) {
        int grid = (int)((n + 255) / 256);
        pad_cvt<<<grid, 256, 0, stream>>>(s, d, srcR, srcC, dstC, n);
    };
    auto pk = [&](const float* s, bf16* d, int srcR, int srcC, int Kd, int Nd) {
        long n = (long)Kd * Nd;
        int grid = (int)((n + 255) / 256);
        pack_b<<<grid, 256, 0, stream>>>(s, d, srcR, srcC, Nd, n);
    };
    auto gemm = [&](const bf16* A, int lda, const bf16* Bpk, int N,
                    const float* bias, float* Cf, bf16* Cbf, int ldc,
                    int M, int K, int flags = 0,
                    const float* gate = nullptr, const float* res = nullptr, int ldr = 0) {
        dim3 grid(N / 64, M / 64), block(128);
        gemm_bf16_wmma<<<grid, block, 0, stream>>>(A, lda, Bpk, N >> 4, bias, Cf, Cbf, ldc,
                                                   K, flags, gate, res, ldr);
    };

    // ------- weight / input conversion + fragment packing -------
    // params leaf order (JAX sorted-key pytree): conv1 base=6, conv2 base=33
    // et: c2p{a,m,p}=+0..2, p2c=+3..5, p2p=+6..8
    // nt.crane: a.W+9 a.b+10 k.W+11 k.b+12 q.W+13 q.b+14 skip+15 v.W+16 v.b+17
    // nt.pile : +18..+26 (same order)
    for (int L = 0; L < 2; ++L) {
        int base = 6 + 27 * L;
        int Ksrc = L ? HID : 88;
        int Kdst = L ? HID : 96;
        for (int t = 0; t < 2; ++t) {
            int o = base + (t ? 18 : 9);
            pk(F(o + 2), Wk[L][t], Ksrc, HID, Kdst, HID);
            pk(F(o + 4), Wq[L][t], Ksrc, HID, Kdst, HID);
            pk(F(o + 7), Wv[L][t], Ksrc, HID, Kdst, HID);
            pk(F(o + 0), Wa[L][t], HID, HID, HID, HID);
        }
        for (int r = 0; r < 3; ++r)
            for (int h = 0; h < NH; ++h) {
                long wo = (long)h * DH * DH;
                pk(F(base + 3 * r + 0) + wo, Arel[L][r] + wo, DH, DH, DH, DH);
                pk(F(base + 3 * r + 1) + wo, Mrel[L][r] + wo, DH, DH, DH, DH);
            }
    }
    pk(F(60), cosWb, NCOS, HID, NCOS, HID);
    pk(F(62), ff1Wb, HID, HID, HID, HID);
    pk(F(64), ff2Wb, HID, ACT, HID, ACT);
    cvt(F(0), Xc[0], NCR, 88, 96, (long)NCR * 96);    // x_crane pad 88->96
    cvt(F(1), Xp[0], NPIL, 88, 96, (long)NPIL * 96);  // x_pile  pad 88->96

    // ------- two HGT conv layers -------
    for (int L = 0; L < 2; ++L) {
        int base = 6 + 27 * L;
        int Kin = L ? HID : 96;
        const float* resC = L ? O1c : nullptr;
        const float* resP = L ? O1p : nullptr;
        float* Oc = L ? O2c : O1c;
        float* Op = L ? O2p : O1p;
        bf16* nxtC = L ? nullptr : Xc[1];
        bf16* nxtP = L ? nullptr : Xp[1];
        int oc = base + 9, op = base + 18;

        // K/Q/V projections (WMMA GEMMs)
        gemm(Xc[L], Kin, Wk[L][0], HID, F(oc + 3), nullptr, Kcb, HID, NCR, Kin);
        gemm(Xc[L], Kin, Wq[L][0], HID, F(oc + 5), Qc, nullptr, HID, NCR, Kin);
        gemm(Xc[L], Kin, Wv[L][0], HID, F(oc + 8), nullptr, Vcb, HID, NCR, Kin);
        gemm(Xp[L], Kin, Wk[L][1], HID, F(op + 3), nullptr, Kpb, HID, NPIL, Kin);
        gemm(Xp[L], Kin, Wq[L][1], HID, F(op + 5), Qp, nullptr, HID, NPIL, Kin);
        gemm(Xp[L], Kin, Wv[L][1], HID, F(op + 8), nullptr, Vpb, HID, NPIL, Kin);

        // per-head relation transforms (hoisted from per-edge to per-node):
        // rel index: 0=c2p (src=crane), 1=p2c (src=pile), 2=p2p (src=pile)
        for (int h = 0; h < NH; ++h) {
            long ho = (long)h * DH;
            long wo = (long)h * DH * DH;
            gemm(Kcb + ho, HID, Arel[L][0] + wo, DH, nullptr, nullptr, KAcp + ho, HID, NCR, DH);
            gemm(Vcb + ho, HID, Mrel[L][0] + wo, DH, nullptr, nullptr, VMcp + ho, HID, NCR, DH);
            gemm(Kpb + ho, HID, Arel[L][1] + wo, DH, nullptr, nullptr, KApc + ho, HID, NPIL, DH);
            gemm(Vpb + ho, HID, Mrel[L][1] + wo, DH, nullptr, nullptr, VMpc + ho, HID, NPIL, DH);
            gemm(Kpb + ho, HID, Arel[L][2] + wo, DH, nullptr, nullptr, KApp + ho, HID, NPIL, DH);
            gemm(Vpb + ho, HID, Mrel[L][2] + wo, DH, nullptr, nullptr, VMpp + ho, HID, NPIL, DH);
        }

        // edge attention + aggregation
        edge_p2c<<<dim3(NCR, NH), 128, 0, stream>>>(Qc, KApc, VMpc, F(base + 5), AGGc);
        edge_pile<<<dim3(NPIL, NH), 128, 0, stream>>>(Qp, KApp, VMpp, VMcp, ei_pp,
                                                      F(base + 8), AGGp);

        // gelu(agg) -> bf16, then output projection (+skip gate on conv2) + SELU
        gelu_cvt<<<(NCR * HID + 255) / 256, 256, 0, stream>>>(AGGc, Gc, (long)NCR * HID);
        gelu_cvt<<<(NPIL * HID + 255) / 256, 256, 0, stream>>>(AGGp, Gp, (long)NPIL * HID);
        int fl = EPI_SELU | (L ? EPI_GATE : 0);
        gemm(Gc, HID, Wa[L][0], HID, F(oc + 1), Oc, nxtC, HID, NCR, HID,
             fl, F(oc + 6), resC, HID);
        gemm(Gp, HID, Wa[L][1], HID, F(op + 1), Op, nxtP, HID, NPIL, HID,
             fl, F(op + 6), resP, HID);
    }

    // ------- IQN quantile head -------
    pool_combine<<<dim3(BATCH, 4), 128, 0, stream>>>(O2p, O2c, hvec);
    cos_feat<<<BATCH * TT, NCOS, 0, stream>>>(taus, cosA);
    gemm(cosA, NCOS, cosWb, HID, F(61), cosx, nullptr, HID, BATCH * TT, NCOS, EPI_SELU);
    mul_hz<<<((long)BATCH * TT * HID + 255) / 256, 256, 0, stream>>>(
        hvec, cosx, zb1, (long)BATCH * TT * HID);
    gemm(zb1, HID, ff1Wb, HID, F(63), nullptr, zb2, HID, BATCH * TT, HID, EPI_SELU);
    gemm(zb2, HID, ff2Wb, ACT, F(65), outq, nullptr, ACT, BATCH * TT, HID);
    copy_f32<<<(BATCH * TT + 255) / 256, 256, 0, stream>>>(
        taus, outq + (long)BATCH * TT * ACT, BATCH * TT);
}
